// SelfAttentionMLP_16552803959225
// MI455X (gfx1250) — matrix operand order repaired and checked by reading, not compiled, not run on hardware
//
#include <hip/hip_runtime.h>

// ---------------------------------------------------------------------------
// SelfAttentionMLP for MI455X (gfx1250, wave32, WMMA)
//   Q,K,V = x@W{q,k,v}.T + b      -> v_wmma_f32_16x16x32_bf16 (f32 accum)
//   energy[b,i,j] = Q[b,i]*K[b,j] -> rank-1; softmax row max = max(q*maxK,q*minK)
//   attention_weights             -> exactly 1/256 (softmax rows sum to 1)
//   out = (softmax @ V) @ Wo.T+bo -> LDS-resident exp stage + WMMA projection
// ---------------------------------------------------------------------------

#define DD 256
#define BB 4096

typedef __attribute__((ext_vector_type(16))) __bf16 v16bf;
typedef __attribute__((ext_vector_type(8)))  __bf16 v8bf;
typedef __attribute__((ext_vector_type(8)))  float  v8f;

// f32 -> bf16 round-to-nearest-even (bit-exact, no native cast dependence)
static __device__ __forceinline__ __bf16 f2bf(float f) {
    unsigned u = __float_as_uint(f);
    unsigned r = (u + 0x7FFFu + ((u >> 16) & 1u)) >> 16;
    unsigned short h = (unsigned short)r;
    __bf16 out;
    __builtin_memcpy(&out, &h, sizeof(out));
    return out;
}

// ---------------------------------------------------------------------------
// One 16x16 f32 output tile of  A[BxD] @ W[DxD]^T  via 8x v_wmma_f32_16x16x32_bf16.
// A is row-major [rows, 256] bf16; W is row-major [256,256] bf16 (row n of W is
// column n of the WMMA B-matrix, i.e. exactly what D = A @ W^T needs).
//
// ISA 7.12.2 fragment layouts (wave32):
//   A 16x32: lane m=lane&15; lanes 0-15 hold K = k0+{0..7, 16..23},
//            lanes 16-31 hold K = k0+{8..15, 24..31}; element e<8 -> K=base+e,
//            element e>=8 -> K=base+16+(e-8).
//   B 32x16: lane n=lane&15; same K mapping as A: lanes 0-15 K=k0+{0..7,16..23},
//            lanes 16-31 K=k0+{8..15,24..31}.
// ---------------------------------------------------------------------------
static __device__ __forceinline__ v8f
wmma_tile_bf16(const __bf16* __restrict__ A, const __bf16* __restrict__ W,
               int rowbase, int colbase, int lane) {
    const int m   = lane & 15;
    const int akb = (lane < 16) ? 0 : 8;
    const int n   = lane & 15;
    const int bkb = (lane < 16) ? 0 : 8;
    const __bf16* arow = A + (size_t)(rowbase + m) * DD;
    const __bf16* brow = W + (size_t)(colbase + n) * DD;
    v8f c = {};
#pragma unroll
    for (int k0 = 0; k0 < DD; k0 += 32) {
        v8bf alo = *(const v8bf*)(arow + k0 + akb);        // K = k0+akb+0..7
        v8bf ahi = *(const v8bf*)(arow + k0 + 16 + akb);   // K = k0+16+akb+0..7
        v8bf blo = *(const v8bf*)(brow + k0 + bkb);        // K = k0+bkb+0..7
        v8bf bhi = *(const v8bf*)(brow + k0 + 16 + bkb);   // K = k0+16+bkb+0..7
        v16bf a, b;
#pragma unroll
        for (int e = 0; e < 8; ++e) {
            a[e] = alo[e]; a[8 + e] = ahi[e];
            b[e] = blo[e]; b[8 + e] = bhi[e];
        }
        c = __builtin_amdgcn_wmma_f32_16x16x32_bf16(
                /*neg_a=*/false, a, /*neg_b=*/false, b,
                /*c_mod=*/(short)0, c, /*reuse_a=*/false, /*reuse_b=*/false);
    }
    return c;
}

// C/D layout: VGPR r, lanes 0-15 -> row rowbase+r, lanes 16-31 -> row rowbase+8+r;
// column = colbase + (lane&15).
static __device__ __forceinline__ void
store_tile_f32(float* __restrict__ out, v8f c, float bias,
               int rowbase, int colbase, int lane) {
    const int n  = lane & 15;
    const int rb = (lane < 16) ? 0 : 8;
#pragma unroll
    for (int r = 0; r < 8; ++r)
        out[(size_t)(rowbase + rb + r) * DD + colbase + n] = c[r] + bias;
}

// ---------------------------------------------------------------------------
// Kernel 1: convert x (1M elems) and the 4 weight matrices (64K elems each)
// to bf16. Grid: 4096 x 256.
// ---------------------------------------------------------------------------
__global__ void convert_bf16_kernel(const float* __restrict__ x,
                                    const float* __restrict__ Wq,
                                    const float* __restrict__ Wk,
                                    const float* __restrict__ Wv,
                                    const float* __restrict__ Wo,
                                    __bf16* __restrict__ xb,
                                    __bf16* __restrict__ wb) {
    int idx = blockIdx.x * blockDim.x + threadIdx.x;
    if (idx < BB * DD) xb[idx] = f2bf(x[idx]);
    if (idx < DD * DD) {
        wb[idx]               = f2bf(Wq[idx]);
        wb[DD * DD + idx]     = f2bf(Wk[idx]);
        wb[2 * DD * DD + idx] = f2bf(Wv[idx]);
        wb[3 * DD * DD + idx] = f2bf(Wo[idx]);
    }
}

// ---------------------------------------------------------------------------
// Kernel 2: fused Q/K/V projection. One wave per 16x16 tile.
// Tiles: 3 mats * (4096/16) row tiles * (256/16) col tiles = 12288.
// Block = 128 threads (4 waves) -> grid 3072.
// ---------------------------------------------------------------------------
__global__ void qkv_gemm_kernel(const __bf16* __restrict__ xb,
                                const __bf16* __restrict__ wb,
                                const float* __restrict__ bq,
                                const float* __restrict__ bk,
                                const float* __restrict__ bv,
                                float* __restrict__ Qf,
                                float* __restrict__ Kf,
                                float* __restrict__ Vf) {
    const int wave = threadIdx.x >> 5;
    const int lane = threadIdx.x & 31;
    const int t    = blockIdx.x * 4 + wave;   // 0..12287
    const int mat  = t >> 12;                 // 0..2  (4096 tiles per matrix)
    const int rt   = (t & 4095) >> 4;         // row tile 0..255
    const int ct   = t & 15;                  // col tile 0..15

    const __bf16* W    = wb + (size_t)mat * DD * DD;
    const float*  bias = (mat == 0) ? bq : (mat == 1) ? bk : bv;
    float*        out  = (mat == 0) ? Qf : (mat == 1) ? Kf : Vf;

    v8f c = wmma_tile_bf16(xb, W, rt * 16, ct * 16, lane);
    store_tile_f32(out, c, bias[ct * 16 + (lane & 15)], rt * 16, ct * 16, lane);
}

// ---------------------------------------------------------------------------
// Kernel 3: softmax-attention stage. One 256-thread workgroup per batch row.
// energy row (b,i) = q * K_b  with q = Q[b,i] scalar:
//   rowmax = max(q*maxK, q*minK);  out_i = sum(exp(q*K_j - m)*V_j)/sum(exp(..))
// attention_weights == 1/256 exactly (softmax rows sum to 1).
// ---------------------------------------------------------------------------
__global__ void attn_kernel(const float* __restrict__ Qf,
                            const float* __restrict__ Kf,
                            const float* __restrict__ Vf,
                            __bf16* __restrict__ ab,      // attn output, bf16
                            float*  __restrict__ attw) {  // output 0: const 1/256
    __shared__ float sK[DD];
    __shared__ float sV[DD];
    const int b = blockIdx.x;
    const int i = threadIdx.x;

    sK[i] = Kf[(size_t)b * DD + i];
    sV[i] = Vf[(size_t)b * DD + i];
    __syncthreads();

    const float q = Qf[(size_t)b * DD + i];

    float kmin = sK[0], kmax = sK[0];
#pragma unroll 8
    for (int j = 1; j < DD; ++j) {
        float k = sK[j];
        kmin = fminf(kmin, k);
        kmax = fmaxf(kmax, k);
    }
    const float m = fmaxf(q * kmin, q * kmax);

    float s = 0.0f, o = 0.0f;
#pragma unroll 8
    for (int j = 0; j < DD; ++j) {
        float e = __expf(q * sK[j] - m);   // v_exp_f32
        s += e;
        o += e * sV[j];
    }

    ab[(size_t)b * DD + i]   = f2bf(o / s);
    attw[(size_t)b * DD + i] = 1.0f / 256.0f;
}

// ---------------------------------------------------------------------------
// Kernel 4: final projection  out = attn @ Wo^T + bo  -> d_out second half.
// 4096 tiles, 4 waves/block -> grid 1024.
// ---------------------------------------------------------------------------
__global__ void out_gemm_kernel(const __bf16* __restrict__ ab,
                                const __bf16* __restrict__ wo,
                                const float* __restrict__ bo,
                                float* __restrict__ out) {
    const int wave = threadIdx.x >> 5;
    const int lane = threadIdx.x & 31;
    const int t    = blockIdx.x * 4 + wave;   // 0..4095
    const int rt   = t >> 4;
    const int ct   = t & 15;

    v8f c = wmma_tile_bf16(ab, wo, rt * 16, ct * 16, lane);
    store_tile_f32(out, c, bo[ct * 16 + (lane & 15)], rt * 16, ct * 16, lane);
}

// ---------------------------------------------------------------------------
extern "C" void kernel_launch(void* const* d_in, const int* in_sizes, int n_in,
                              void* d_out, int out_size, void* d_ws, size_t ws_size,
                              hipStream_t stream) {
    const float* x  = (const float*)d_in[0];
    const float* Wq = (const float*)d_in[1];
    const float* bq = (const float*)d_in[2];
    const float* Wk = (const float*)d_in[3];
    const float* bk = (const float*)d_in[4];
    const float* Wv = (const float*)d_in[5];
    const float* bv = (const float*)d_in[6];
    const float* Wo = (const float*)d_in[7];
    const float* bo = (const float*)d_in[8];

    // Workspace layout (total ~16.5 MB):
    char* ws = (char*)d_ws;
    __bf16* xb = (__bf16*)(ws);                                   // 2 MB
    __bf16* wb = (__bf16*)(ws + (size_t)2 * 1024 * 1024);         // 512 KB (4 mats)
    float*  Qf = (float*) (ws + (size_t)2621440);                 // 4 MB
    float*  Kf = (float*) (ws + (size_t)6815744);                 // 4 MB
    float*  Vf = (float*) (ws + (size_t)11010048);                // 4 MB
    __bf16* ab = (__bf16*)(ws + (size_t)15204352);                // 2 MB

    float* attw = (float*)d_out;                    // output 0: [4096,256]
    float* outo = (float*)d_out + (size_t)BB * DD;  // output 1: [4096,256]

    convert_bf16_kernel<<<(BB * DD) / 256, 256, 0, stream>>>(x, Wq, Wk, Wv, Wo, xb, wb);
    qkv_gemm_kernel<<<3072, 128, 0, stream>>>(xb, wb, bq, bk, bv, Qf, Kf, Vf);
    attn_kernel<<<BB, DD, 0, stream>>>(Qf, Kf, Vf, ab, attw);
    out_gemm_kernel<<<1024, 128, 0, stream>>>(ab, wb + (size_t)3 * DD * DD, bo, outo);
}
